// MultiHeadAttention_36077725286739
// MI455X (gfx1250) — compile-verified, matrix operand order checked by reading, not run on hardware
//
#include <hip/hip_runtime.h>
#include <hip/hip_bf16.h>

// ---------------------------------------------------------------------------
// MHA forward for B=4, T=2048, C=1024, H=16, D=64 on gfx1250 (MI455X).
// All matmuls via v_wmma_f32_16x16x32_bf16 (bf16 in, fp32 accum).
// Round 2: 32x64 wave tiles + software pipelining (double-buffered fragment
// loads) in the GEMMs; prefetched K/V fragments + exp2-domain softmax in the
// flash-attention kernel; vectorized bf16 downconverts.
// ---------------------------------------------------------------------------

typedef __bf16 bf16_t;
typedef __attribute__((ext_vector_type(16))) __bf16 v16bf;
typedef __attribute__((ext_vector_type(8)))  __bf16 v8bf;
typedef __attribute__((ext_vector_type(8)))  float  v8f;

#define BB 4
#define TT 2048
#define CC 1024
#define HH 16
#define DD 64
#define C3 3072

__device__ inline v8f wmma_bf16(v16bf a, v16bf b, v8f c) {
    return __builtin_amdgcn_wmma_f32_16x16x32_bf16(
        /*neg_a=*/false, a, /*neg_b=*/false, b,
        /*c_mod=*/(short)0, c, /*reuse_a=*/false, /*reuse_b=*/false);
}

// Load a 16-bit 16x32 A/B fragment for this lane.
// CDNA5 layout: lane's 16 elements are K = grp*8+0..7 and K = 16+grp*8+0..7,
// i.e. two contiguous 16-byte runs -> two b128 loads.
// Caller passes p = row_base + grp*8 (16-byte aligned).
__device__ inline v16bf ldfrag(const bf16_t* p) {
    v16bf f;
    ((v8bf*)&f)[0] = *(const v8bf*)(p);
    ((v8bf*)&f)[1] = *(const v8bf*)(p + 16);
    return f;
}

// 8 WMMAs for a 32(M) x 64(N) x 32(K) step.
__device__ inline void mm8(v8f acc[8], v16bf a0, v16bf a1,
                           v16bf b0, v16bf b1, v16bf b2, v16bf b3) {
    acc[0] = wmma_bf16(a0, b0, acc[0]);
    acc[1] = wmma_bf16(a0, b1, acc[1]);
    acc[2] = wmma_bf16(a0, b2, acc[2]);
    acc[3] = wmma_bf16(a0, b3, acc[3]);
    acc[4] = wmma_bf16(a1, b0, acc[4]);
    acc[5] = wmma_bf16(a1, b1, acc[5]);
    acc[6] = wmma_bf16(a1, b2, acc[6]);
    acc[7] = wmma_bf16(a1, b3, acc[7]);
}

// ---------------------------------------------------------------------------
// 1) fp32 -> bf16 conversion, 8 elems/iter (2x b128 load, 1x b128 store)
// ---------------------------------------------------------------------------
__global__ __launch_bounds__(256) void cvt_bf16_kernel(const float* __restrict__ in,
                                                       bf16_t* __restrict__ out, int n8) {
    int i = blockIdx.x * blockDim.x + threadIdx.x;
    int stride = gridDim.x * blockDim.x;
    for (; i < n8; i += stride) {
        const float4* p = (const float4*)in + 2 * (size_t)i;
        float4 u = p[0], w = p[1];
        v8bf o;
        o[0] = (bf16_t)u.x; o[1] = (bf16_t)u.y; o[2] = (bf16_t)u.z; o[3] = (bf16_t)u.w;
        o[4] = (bf16_t)w.x; o[5] = (bf16_t)w.y; o[6] = (bf16_t)w.z; o[7] = (bf16_t)w.w;
        ((v8bf*)out)[i] = o;
    }
}

// ---------------------------------------------------------------------------
// 2) QKV projection: qkv[bt,o] = sum_c x[bt,c] * W_attn[o,c]
//    M=8192, N=3072, K=1024. Wave tile = 32(M) x 64(N), double-buffered.
// ---------------------------------------------------------------------------
__global__ __launch_bounds__(256) void gemm_qkv_kernel(const bf16_t* __restrict__ xb,
                                                       const bf16_t* __restrict__ wab,
                                                       bf16_t* __restrict__ qkvb) {
    const int lane = threadIdx.x & 31;
    const int wave = blockIdx.x * (blockDim.x >> 5) + (threadIdx.x >> 5);
    const int numN = C3 / 64;              // 48
    const int wn = wave % numN;
    const int wm = wave / numN;            // < 256
    const int m0 = wm * 32, n0 = wn * 64;
    const int ln = lane & 15, grp = lane >> 4;

    v8f acc[8] = {};
    const bf16_t* arow0 = xb + (size_t)(m0 + ln) * CC + grp * 8;
    const bf16_t* arow1 = arow0 + (size_t)16 * CC;
    const bf16_t* brow  = wab + (size_t)(n0 + ln) * CC + grp * 8;

    v16bf a0 = ldfrag(arow0), a1 = ldfrag(arow1);
    v16bf b0 = ldfrag(brow);
    v16bf b1 = ldfrag(brow + (size_t)16 * CC);
    v16bf b2 = ldfrag(brow + (size_t)32 * CC);
    v16bf b3 = ldfrag(brow + (size_t)48 * CC);

    for (int kb = 32; kb < CC; kb += 32) {
        v16bf na0 = ldfrag(arow0 + kb), na1 = ldfrag(arow1 + kb);
        v16bf nb0 = ldfrag(brow + kb);
        v16bf nb1 = ldfrag(brow + (size_t)16 * CC + kb);
        v16bf nb2 = ldfrag(brow + (size_t)32 * CC + kb);
        v16bf nb3 = ldfrag(brow + (size_t)48 * CC + kb);
        mm8(acc, a0, a1, b0, b1, b2, b3);
        a0 = na0; a1 = na1; b0 = nb0; b1 = nb1; b2 = nb2; b3 = nb3;
    }
    mm8(acc, a0, a1, b0, b1, b2, b3);

#pragma unroll
    for (int mi = 0; mi < 2; mi++) {
#pragma unroll
        for (int j = 0; j < 4; j++) {
#pragma unroll
            for (int v = 0; v < 8; v++) {
                int row = m0 + mi * 16 + v + grp * 8;
                int col = n0 + j * 16 + ln;
                qkvb[(size_t)row * C3 + col] = (bf16_t)acc[mi * 4 + j][v];
            }
        }
    }
}

// ---------------------------------------------------------------------------
// 3) V transpose: vT[((b*H+h)*D+d)*T + t] = qkv[b*T+t, 2C + h*D + d]
// ---------------------------------------------------------------------------
__global__ __launch_bounds__(256) void transpose_v_kernel(const bf16_t* __restrict__ qkvb,
                                                          bf16_t* __restrict__ vTb) {
    int i = blockIdx.x * blockDim.x + threadIdx.x;   // over B*H*D*T = 8M
    if (i >= BB * HH * DD * TT) return;
    int t = i & (TT - 1);
    int d = (i >> 11) & (DD - 1);
    int h = (i >> 17) & (HH - 1);
    int b = i >> 21;
    vTb[i] = qkvb[(size_t)(b * TT + t) * C3 + 2 * CC + h * DD + d];
}

// ---------------------------------------------------------------------------
// 4) Flash attention. One wave = 16 query rows of one (b,h).
//    Per 32-key block: 4 WMMA for S (K-dim d=64), online softmax in the exp2
//    domain, P bounced via wave-private LDS, 4 WMMA for O += P V.
//    K-fragments of the NEXT block and V-fragments of the CURRENT block are
//    issued before the S WMMAs so global latency hides under softmax VALU.
// ---------------------------------------------------------------------------
__global__ __launch_bounds__(256) void attn_kernel(const bf16_t* __restrict__ qkvb,
                                                   const bf16_t* __restrict__ vTb,
                                                   bf16_t* __restrict__ yb) {
    __shared__ bf16_t plds[8 * 512];       // 16x32 bf16 per wave, 8 waves
    const int lane = threadIdx.x & 31;
    const int wslot = threadIdx.x >> 5;
    const int wave = blockIdx.x * 8 + wslot;
    const int qi = wave & (TT / 16 - 1);   // 0..127
    const int h  = (wave >> 7) & (HH - 1);
    const int b  = wave >> 11;
    const int ln = lane & 15, grp = lane >> 4;
    // fold 1/sqrt(D) and log2(e) into one scale; softmax runs in exp2 domain
    const float scale2 = 0.125f * 1.4426950408889634f;

    // Q fragments (kept resident): 16 rows x 64 d -> two 16x32 frags
    const bf16_t* qp = qkvb + (size_t)(b * TT + qi * 16 + ln) * C3 + h * DD;
    v16bf qlo = ldfrag(qp + grp * 8);
    v16bf qhi = ldfrag(qp + 32 + grp * 8);

    const bf16_t* vbase = vTb + (size_t)((b * HH + h) * DD) * TT + grp * 8;
    // key-fragment base for this lane: row (b*T + key), col C + h*D
    const bf16_t* kbase = qkvb + (size_t)(b * TT + ln) * C3 + CC + h * DD + grp * 8;
    bf16_t* pst = plds + wslot * 512;

    float mrow[8], lrow[8];
    v8f O[4] = {};
#pragma unroll
    for (int v = 0; v < 8; v++) { mrow[v] = -1e30f; lrow[v] = 0.0f; }

    // preload K fragments for key block 0
    v16bf k0lo = ldfrag(kbase);
    v16bf k0hi = ldfrag(kbase + 32);
    v16bf k1lo = ldfrag(kbase + (size_t)16 * C3);
    v16bf k1hi = ldfrag(kbase + (size_t)16 * C3 + 32);

    for (int t0 = 0; t0 < TT; t0 += 32) {
        // prefetch next key block (wraps to 0 on last iter -> branch-free)
        const int tn = (t0 + 32) & (TT - 1);
        const bf16_t* kn = kbase + (size_t)tn * C3;
        v16bf nk0lo = ldfrag(kn);
        v16bf nk0hi = ldfrag(kn + 32);
        v16bf nk1lo = ldfrag(kn + (size_t)16 * C3);
        v16bf nk1hi = ldfrag(kn + (size_t)16 * C3 + 32);
        // prefetch V fragments for the current block
        v16bf Vf0 = ldfrag(vbase + (size_t)(0 * 16 + ln) * TT + t0);
        v16bf Vf1 = ldfrag(vbase + (size_t)(1 * 16 + ln) * TT + t0);
        v16bf Vf2 = ldfrag(vbase + (size_t)(2 * 16 + ln) * TT + t0);
        v16bf Vf3 = ldfrag(vbase + (size_t)(3 * 16 + ln) * TT + t0);

        // ---- S = Q K^T for 32 keys (two 16x16 tiles) ----
        v8f S0 = {}, S1 = {};
        S0 = wmma_bf16(qlo, k0lo, S0);
        S0 = wmma_bf16(qhi, k0hi, S0);
        S1 = wmma_bf16(qlo, k1lo, S1);
        S1 = wmma_bf16(qhi, k1hi, S1);

        // ---- online softmax update (row m = v + 8*grp, col = key) ----
#pragma unroll
        for (int v = 0; v < 8; v++) {
            float s0 = S0[v] * scale2;
            float s1 = S1[v] * scale2;
            float rm = fmaxf(s0, s1);
            rm = fmaxf(rm, __shfl_xor(rm, 1, 32));
            rm = fmaxf(rm, __shfl_xor(rm, 2, 32));
            rm = fmaxf(rm, __shfl_xor(rm, 4, 32));
            rm = fmaxf(rm, __shfl_xor(rm, 8, 32));
            float mn = fmaxf(mrow[v], rm);
            float alpha = __builtin_amdgcn_exp2f(mrow[v] - mn);
            float p0 = __builtin_amdgcn_exp2f(s0 - mn);
            float p1 = __builtin_amdgcn_exp2f(s1 - mn);
            float rs = p0 + p1;
            rs += __shfl_xor(rs, 1, 32);
            rs += __shfl_xor(rs, 2, 32);
            rs += __shfl_xor(rs, 4, 32);
            rs += __shfl_xor(rs, 8, 32);
            lrow[v] = lrow[v] * alpha + rs;
            mrow[v] = mn;
#pragma unroll
            for (int j = 0; j < 4; j++) O[j][v] *= alpha;
            // stage P (16x32 row-major bf16) into wave-private LDS
            int r = v + grp * 8;
            pst[r * 32 + ln]      = (bf16_t)p0;
            pst[r * 32 + 16 + ln] = (bf16_t)p1;
        }
        // reload P in A-fragment layout (2x ds_load_b128; same-wave DS ordered)
        v16bf P = ldfrag(pst + ln * 32 + grp * 8);
        // ---- O += P V ----
        O[0] = wmma_bf16(P, Vf0, O[0]);
        O[1] = wmma_bf16(P, Vf1, O[1]);
        O[2] = wmma_bf16(P, Vf2, O[2]);
        O[3] = wmma_bf16(P, Vf3, O[3]);

        k0lo = nk0lo; k0hi = nk0hi; k1lo = nk1lo; k1hi = nk1hi;
    }
    // ---- finalize: y[b, t, h*64+d] = O / l ----
#pragma unroll
    for (int v = 0; v < 8; v++) {
        float inv = 1.0f / lrow[v];
        int trow = qi * 16 + v + grp * 8;
        bf16_t* yp = yb + (size_t)(b * TT + trow) * CC + h * DD;
#pragma unroll
        for (int j = 0; j < 4; j++)
            yp[j * 16 + ln] = (bf16_t)(O[j][v] * inv);
    }
}

// ---------------------------------------------------------------------------
// 5) Output projection: out[bt,o] = sum_c y[bt,c]*W_proj[o,c] + b_proj[o]
//    M=8192, N=1024, K=1024. Wave tile 32x64, double-buffered, fp32 stores.
// ---------------------------------------------------------------------------
__global__ __launch_bounds__(256) void gemm_proj_kernel(const bf16_t* __restrict__ yb,
                                                        const bf16_t* __restrict__ wpb,
                                                        const float* __restrict__ bias,
                                                        float* __restrict__ out) {
    const int lane = threadIdx.x & 31;
    const int wave = blockIdx.x * (blockDim.x >> 5) + (threadIdx.x >> 5);
    const int numN = CC / 64;              // 16
    const int wn = wave % numN;
    const int wm = wave / numN;            // < 256
    const int m0 = wm * 32, n0 = wn * 64;
    const int ln = lane & 15, grp = lane >> 4;

    v8f acc[8] = {};
    const bf16_t* arow0 = yb + (size_t)(m0 + ln) * CC + grp * 8;
    const bf16_t* arow1 = arow0 + (size_t)16 * CC;
    const bf16_t* brow  = wpb + (size_t)(n0 + ln) * CC + grp * 8;

    v16bf a0 = ldfrag(arow0), a1 = ldfrag(arow1);
    v16bf b0 = ldfrag(brow);
    v16bf b1 = ldfrag(brow + (size_t)16 * CC);
    v16bf b2 = ldfrag(brow + (size_t)32 * CC);
    v16bf b3 = ldfrag(brow + (size_t)48 * CC);

    for (int kb = 32; kb < CC; kb += 32) {
        v16bf na0 = ldfrag(arow0 + kb), na1 = ldfrag(arow1 + kb);
        v16bf nb0 = ldfrag(brow + kb);
        v16bf nb1 = ldfrag(brow + (size_t)16 * CC + kb);
        v16bf nb2 = ldfrag(brow + (size_t)32 * CC + kb);
        v16bf nb3 = ldfrag(brow + (size_t)48 * CC + kb);
        mm8(acc, a0, a1, b0, b1, b2, b3);
        a0 = na0; a1 = na1; b0 = nb0; b1 = nb1; b2 = nb2; b3 = nb3;
    }
    mm8(acc, a0, a1, b0, b1, b2, b3);

#pragma unroll
    for (int mi = 0; mi < 2; mi++) {
#pragma unroll
        for (int j = 0; j < 4; j++) {
#pragma unroll
            for (int v = 0; v < 8; v++) {
                int row = m0 + mi * 16 + v + grp * 8;
                int col = n0 + j * 16 + ln;
                out[(size_t)row * CC + col] = acc[mi * 4 + j][v] + bias[col];
            }
        }
    }
}

// ---------------------------------------------------------------------------
// Launch
// ---------------------------------------------------------------------------
extern "C" void kernel_launch(void* const* d_in, const int* in_sizes, int n_in,
                              void* d_out, int out_size, void* d_ws, size_t ws_size,
                              hipStream_t stream) {
    const float* x      = (const float*)d_in[0];   // [B,T,C]
    const float* W_attn = (const float*)d_in[1];   // [3C,C]
    const float* W_proj = (const float*)d_in[2];   // [C,C]
    const float* b_proj = (const float*)d_in[3];   // [C]
    float* out = (float*)d_out;

    const size_t n_x  = (size_t)BB * TT * CC;      // 8,388,608
    const size_t n_wa = (size_t)C3 * CC;           // 3,145,728
    const size_t n_wp = (size_t)CC * CC;           // 1,048,576
    const size_t n_qkv = (size_t)BB * TT * C3;     // 25,165,824
    const size_t n_vT  = (size_t)BB * HH * DD * TT;// 8,388,608

    char* w = (char*)d_ws;
    bf16_t* xb   = (bf16_t*)w; w += n_x   * sizeof(bf16_t);
    bf16_t* wab  = (bf16_t*)w; w += n_wa  * sizeof(bf16_t);
    bf16_t* wpb  = (bf16_t*)w; w += n_wp  * sizeof(bf16_t);
    bf16_t* qkvb = (bf16_t*)w; w += n_qkv * sizeof(bf16_t);
    bf16_t* vTb  = (bf16_t*)w; w += n_vT  * sizeof(bf16_t);
    bf16_t* yb   = (bf16_t*)w; w += n_x   * sizeof(bf16_t);

    // 1-3: bf16 downconvert (8 elements per thread-iteration)
    cvt_bf16_kernel<<<2048, 256, 0, stream>>>(x, xb, (int)(n_x / 8));
    cvt_bf16_kernel<<<1024, 256, 0, stream>>>(W_attn, wab, (int)(n_wa / 8));
    cvt_bf16_kernel<<<512, 256, 0, stream>>>(W_proj, wpb, (int)(n_wp / 8));

    // 4: QKV GEMM — 256*48 = 12288 waves, 8 waves/block
    gemm_qkv_kernel<<<1536, 256, 0, stream>>>(xb, wab, qkvb);

    // 5: V transpose
    transpose_v_kernel<<<(int)(n_vT / 256), 256, 0, stream>>>(qkvb, vTb);

    // 6: flash attention — B*H*(T/16) = 8192 waves, 8 waves/block
    attn_kernel<<<1024, 256, 0, stream>>>(qkvb, vTb, yb);

    // 7: output projection — 256*16 = 4096 waves
    gemm_proj_kernel<<<512, 256, 0, stream>>>(yb, wpb, b_proj, out);
}
